// TinyRMSNormQuantFP8_11218454577402
// MI455X (gfx1250) — 8-bit matrix operand order repaired and checked by reading, compiled, not run on hardware
//
#include <hip/hip_runtime.h>
#include <hip/hip_bf16.h>

#define TOKENS 8192
#define HIDDEN 4096
#define OUTS   4096
#define EPS    1e-5f
#define FP8MAX 448.0f

typedef __attribute__((ext_vector_type(16))) int   v16i;
typedef __attribute__((ext_vector_type(8)))  float v8f;

#if __has_builtin(__builtin_amdgcn_tensor_load_to_lds)
#define USE_TDM 1
#else
#define USE_TDM 0
#endif

__device__ __forceinline__ float bf16lo_to_f32(unsigned u) {
    return __uint_as_float(u << 16);
}
__device__ __forceinline__ float bf16hi_to_f32(unsigned u) {
    return __uint_as_float(u & 0xFFFF0000u);
}
__device__ __forceinline__ unsigned short f32_to_bf16_rne(float f) {
    unsigned u = __float_as_uint(f);
    unsigned r = u + 0x7FFFu + ((u >> 16) & 1u);
    return (unsigned short)(r >> 16);
}

__device__ __forceinline__ unsigned cvt_f32_e4m3(float x) {
    unsigned ub = __float_as_uint(x);
    unsigned s  = (ub >> 24) & 0x80u;
    float a = __uint_as_float(ub & 0x7FFFFFFFu);
    if (!(a < FP8MAX)) a = FP8MAX;
    if (a < 0.015625f) {
        int m = __float2int_rn(a * 512.0f);
        return s | (unsigned)m;
    }
    int E = (int)((__float_as_uint(a) >> 23) & 0xFFu) - 127;
    int m = __float2int_rn(ldexpf(a, 3 - E));
    if (m == 16) { m = 8; ++E; }
    return s | (unsigned)(((E + 7) << 3) | (m & 7));
}

__device__ __forceinline__ void async_copy_b128(unsigned lds_off, const void* gaddr) {
    asm volatile("global_load_async_to_lds_b128 %0, %1, off"
                 :: "v"(lds_off), "v"(gaddr) : "memory");
}
__device__ __forceinline__ void wait_asynccnt_le6() {
#if __has_builtin(__builtin_amdgcn_s_wait_asynccnt)
    __builtin_amdgcn_s_wait_asynccnt(6);
#else
    asm volatile("s_wait_asynccnt 0x6" ::: "memory");
#endif
}
__device__ __forceinline__ void wait_asynccnt_le0() {
#if __has_builtin(__builtin_amdgcn_s_wait_asynccnt)
    __builtin_amdgcn_s_wait_asynccnt(0);
#else
    asm volatile("s_wait_asynccnt 0x0" ::: "memory");
#endif
}
__device__ __forceinline__ unsigned lds_offset_of(const void* p) {
    return (unsigned)(unsigned long long)p;
}

#if USE_TDM
typedef __attribute__((ext_vector_type(4))) unsigned tg0_t;
typedef __attribute__((ext_vector_type(8))) int      tg1_t;
typedef __attribute__((ext_vector_type(4))) int      tg2_t;
typedef __attribute__((ext_vector_type(8))) int      tg3x_t;

__device__ __forceinline__ void tdm_issue(tg0_t g0, tg1_t g1) {
    tg2_t z4 = (tg2_t)0;
#if defined(__clang_major__) && (__clang_major__ >= 23)
    __builtin_amdgcn_tensor_load_to_lds(g0, g1, z4, z4, (tg3x_t)0, 0);
#else
    __builtin_amdgcn_tensor_load_to_lds(g0, g1, z4, z4, 0);
#endif
}

__device__ __forceinline__ void tdm_load_tile(unsigned lds_addr,
                                              const void* gtile,
                                              unsigned rows) {
    const unsigned long long ga = (unsigned long long)gtile;
    tg0_t g0;
    g0[0] = 1u;
    g0[1] = lds_addr;
    g0[2] = (unsigned)ga;
    g0[3] = (unsigned)((ga >> 32) & 0x01FFFFFFu)
          | 0x80000000u;
    tg1_t g1;
    g1[0] = (int)((1u << 20) |
                  (4u << 22) |
                  (3u << 25));
    g1[1] = (int)((unsigned)HIDDEN << 16);
    g1[2] = (int)(rows << 16);
    g1[3] = (int)(128u << 16);
    g1[4] = (int)rows;
    g1[5] = HIDDEN;
    g1[6] = 0;
    g1[7] = 0;
    tdm_issue(g0, g1);
}

__device__ __forceinline__ void wait_tensorcnt_le2() {
    __builtin_amdgcn_s_wait_tensorcnt(2);
}
__device__ __forceinline__ void wait_tensorcnt_le0() {
    __builtin_amdgcn_s_wait_tensorcnt(0);
}
#endif

__global__ void __launch_bounds__(256)
rmsnorm_quant_fp8_kernel(const __hip_bfloat16* __restrict__ x,
                         const __hip_bfloat16* __restrict__ nw,
                         const float* __restrict__ input_scale,
                         unsigned char* __restrict__ q) {
    const int t   = blockIdx.x;
    const int tid = threadIdx.x;

    const uint4* xrow = reinterpret_cast<const uint4*>(x + (size_t)t * HIDDEN);
    const uint4* wrow = reinterpret_cast<const uint4*>(nw);

    uint4 u0 = xrow[tid * 2 + 0];
    uint4 u1 = xrow[tid * 2 + 1];

    float f[16];
    {
        unsigned uu[8] = {u0.x, u0.y, u0.z, u0.w, u1.x, u1.y, u1.z, u1.w};
        #pragma unroll
        for (int i = 0; i < 8; ++i) {
            f[2 * i + 0] = bf16lo_to_f32(uu[i]);
            f[2 * i + 1] = bf16hi_to_f32(uu[i]);
        }
    }

    float ss = 0.0f;
    #pragma unroll
    for (int i = 0; i < 16; ++i) ss += f[i] * f[i];

    #pragma unroll
    for (int o = 16; o > 0; o >>= 1) ss += __shfl_xor(ss, o, 32);

    __shared__ float red[8];
    __shared__ float s_invrms;
    const int wave = tid >> 5;
    const int lane = tid & 31;
    if (lane == 0) red[wave] = ss;
    __syncthreads();
    if (tid == 0) {
        float tot = 0.0f;
        #pragma unroll
        for (int i = 0; i < 8; ++i) tot += red[i];
        s_invrms = rsqrtf(tot * (1.0f / (float)HIDDEN) + EPS);
    }
    __syncthreads();
    const float inv_rms = s_invrms;
    const float inv_is  = 1.0f / input_scale[0];

    uint4 w0 = wrow[tid * 2 + 0];
    uint4 w1 = wrow[tid * 2 + 1];
    float wf[16];
    {
        unsigned uu[8] = {w0.x, w0.y, w0.z, w0.w, w1.x, w1.y, w1.z, w1.w};
        #pragma unroll
        for (int i = 0; i < 8; ++i) {
            wf[2 * i + 0] = bf16lo_to_f32(uu[i]);
            wf[2 * i + 1] = bf16hi_to_f32(uu[i]);
        }
    }

    unsigned d[4];
    #pragma unroll
    for (int j = 0; j < 4; ++j) {
        unsigned pk = 0;
        #pragma unroll
        for (int b = 0; b < 4; ++b) {
            float v = f[4 * j + b] * inv_rms * wf[4 * j + b] * inv_is;
            v = fminf(fmaxf(v, -FP8MAX), FP8MAX);
            pk |= cvt_f32_e4m3(v) << (8 * b);
        }
        d[j] = pk;
    }
    uint4 st; st.x = d[0]; st.y = d[1]; st.z = d[2]; st.w = d[3];
    reinterpret_cast<uint4*>(q + (size_t)t * HIDDEN)[tid] = st;
}

#define KSTEP   128
#define NKITER  (HIDDEN / KSTEP)
#define LPITCH  144

union AFrag { v16i v; unsigned long long u64[8]; };
union BFrag { v16i v; uint4 u128[4]; };

__global__ void __launch_bounds__(256)
fp8_gemm_wmma_kernel(const unsigned char* __restrict__ q,
                     const unsigned char* __restrict__ w,
                     const float* __restrict__ input_scale,
                     const float* __restrict__ weight_scale,
                     __hip_bfloat16* __restrict__ out) {
    __shared__ unsigned char sA[2][64 * LPITCH];
    __shared__ unsigned char sB[2][128 * LPITCH];

    const int tid   = threadIdx.x;
    const int lane  = tid & 31;
    const int wave  = tid >> 5;
    const int waveM = wave >> 2;
    const int waveN = wave & 3;

    const int mBlk = blockIdx.x * 64;
    const int nBlk = blockIdx.y * 128;

    const unsigned char* qg = q + (size_t)mBlk * HIDDEN;
    const unsigned char* wg = w + (size_t)nBlk * HIDDEN;

    const unsigned aL[2] = { lds_offset_of(&sA[0][0]), lds_offset_of(&sA[1][0]) };
    const unsigned bL[2] = { lds_offset_of(&sB[0][0]), lds_offset_of(&sB[1][0]) };

    const int lrow = lane & 15;
    const int hi   = lane >> 4;

    v8f acc[2][2];
    #pragma unroll
    for (int mi = 0; mi < 2; ++mi)
        #pragma unroll
        for (int ni = 0; ni < 2; ++ni)
            acc[mi][ni] = (v8f)(0.0f);

#if USE_TDM
    auto stage = [&](int buf, int k0) {
        tdm_load_tile(aL[buf], qg + k0, 64u);
        tdm_load_tile(bL[buf], wg + k0, 128u);
    };
    if (wave == 0) stage(0, 0);
#else
    const int srow = tid >> 3;
    const int skc  = (tid & 7) << 4;
    auto stage = [&](int buf, int k0) {
        #pragma unroll
        for (int it = 0; it < 2; ++it) {
            int row = srow + it * 32;
            async_copy_b128(aL[buf] + row * LPITCH + skc,
                            qg + (size_t)row * HIDDEN + k0 + skc);
        }
        #pragma unroll
        for (int it = 0; it < 4; ++it) {
            int row = srow + it * 32;
            async_copy_b128(bL[buf] + row * LPITCH + skc,
                            wg + (size_t)row * HIDDEN + k0 + skc);
        }
    };
    stage(0, 0);
#endif

    for (int kk = 0; kk < NKITER; ++kk) {
        const int cur = kk & 1;
#if USE_TDM
        if (wave == 0) {
            if (kk + 1 < NKITER) {
                stage(cur ^ 1, (kk + 1) * KSTEP);
                wait_tensorcnt_le2();
            } else {
                wait_tensorcnt_le0();
            }
        }
#else
        if (kk + 1 < NKITER) {
            stage(cur ^ 1, (kk + 1) * KSTEP);
            wait_asynccnt_le6();
        } else {
            wait_asynccnt_le0();
        }
#endif
        __syncthreads();

        AFrag a[2];
        BFrag b[2];
        #pragma unroll
        for (int mi = 0; mi < 2; ++mi) {
            const unsigned char* p =
                &sA[cur][(waveM * 32 + mi * 16 + lrow) * LPITCH + 16 * hi];
            #pragma unroll
            for (int i = 0; i < 8; ++i)
                a[mi].u64[i] = *reinterpret_cast<const unsigned long long*>(p + 32 * (i >> 1) + 8 * (i & 1));
        }
        #pragma unroll
        for (int ni = 0; ni < 2; ++ni) {
            const unsigned char* p =
                &sB[cur][(waveN * 32 + ni * 16 + lrow) * LPITCH + 16 * hi];
            #pragma unroll
            for (int j = 0; j < 4; ++j)
                b[ni].u128[j] = *reinterpret_cast<const uint4*>(p + 32 * j);
        }

        #pragma unroll
        for (int mi = 0; mi < 2; ++mi)
            #pragma unroll
            for (int ni = 0; ni < 2; ++ni)
                acc[mi][ni] = __builtin_amdgcn_wmma_f32_16x16x128_fp8_fp8(
                    a[mi].v, b[ni].v, (short)0, acc[mi][ni],
                     false,  false);

        __syncthreads();
    }

    const float sc = input_scale[0] * weight_scale[0];

    #pragma unroll
    for (int mi = 0; mi < 2; ++mi) {
        #pragma unroll
        for (int ni = 0; ni < 2; ++ni) {
            const int col  = nBlk + waveN * 32 + ni * 16 + (lane & 15);
            const int row0 = mBlk + waveM * 32 + mi * 16 + ((lane >> 4) << 3);
            #pragma unroll
            for (int r = 0; r < 8; ++r) {
                unsigned short h = f32_to_bf16_rne(acc[mi][ni][r] * sc);
                reinterpret_cast<unsigned short*>(out)[(size_t)(row0 + r) * OUTS + col] = h;
            }
        }
    }
}

extern "C" void kernel_launch(void* const* d_in, const int* in_sizes, int n_in,
                              void* d_out, int out_size, void* d_ws, size_t ws_size,
                              hipStream_t stream) {
    const __hip_bfloat16* x   = (const __hip_bfloat16*)d_in[0];
    const __hip_bfloat16* nw  = (const __hip_bfloat16*)d_in[1];
    const unsigned char*  wq  = (const unsigned char*)d_in[2];
    const float* in_scale     = (const float*)d_in[3];
    const float* w_scale      = (const float*)d_in[4];
    __hip_bfloat16* out       = (__hip_bfloat16*)d_out;
    unsigned char* qact       = (unsigned char*)d_ws;

    rmsnorm_quant_fp8_kernel<<<TOKENS, 256, 0, stream>>>(x, nw, in_scale, qact);

    dim3 grid(TOKENS / 64, OUTS / 128);
    fp8_gemm_wmma_kernel<<<grid, 256, 0, stream>>>(qact, wq, in_scale, w_scale, out);
}
